// SelfAttn_86741159510393
// MI455X (gfx1250) — matrix operand order repaired and checked by reading, not compiled, not run on hardware
//
#include <hip/hip_runtime.h>

// ---------- CDNA5 WMMA types ----------
typedef __attribute__((ext_vector_type(16))) __bf16        v16bf;
typedef __attribute__((ext_vector_type(8)))  float         v8f;
typedef __attribute__((ext_vector_type(4)))  unsigned int  v4u;

union BFrag {
  v16bf          v;
  unsigned int   u[8];
  unsigned short s[16];
  v4u            q[2];
};

// float -> bf16 round-to-nearest-even (bit-level)
__device__ __forceinline__ unsigned short f2bf(float f) {
  unsigned int u = __builtin_bit_cast(unsigned int, f);
  u += 0x7FFFu + ((u >> 16) & 1u);
  return (unsigned short)(u >> 16);
}

// Two f32 -> packed pair of bf16 (RNE), merged with one v_perm_b32.
__device__ __forceinline__ unsigned int f2bf_pk(float x0, float x1) {
  unsigned int u0 = __builtin_bit_cast(unsigned int, x0);
  unsigned int u1 = __builtin_bit_cast(unsigned int, x1);
  u0 += 0x7FFFu + ((u0 >> 16) & 1u);
  u1 += 0x7FFFu + ((u1 >> 16) & 1u);
  // result = [u1.hi16 | u0.hi16] : low half = x0, high half = x1
  return __builtin_amdgcn_perm(u1, u0, 0x07060302u);
}

// A-matrix 16-bit layout (ISA 7.12.2): VGPR pair j holds K = kbase+2j (j<4)
// or 16+kbase+2(j-4) (j>=4), kbase = 8*(lane>=16). Pairs are K,K+1.
__device__ __forceinline__ int a_pair_k(int j, int kbase) {
  return (j < 4) ? (kbase + 2 * j) : (16 + kbase + 2 * (j - 4));
}

#define WPB 4  // waves per block (wave32)

// =====================================================================
// Kernel 0: one-time weight swizzle.  W[256,256] f32 (q,k,v) -> bf16 in
// B-fragment order Wf[w][c][nt][lane][16] so kernel 1 loads fragments as
// two contiguous b128 per lane with zero conversion VALU.
// =====================================================================
__global__ __launch_bounds__(256) void wswz_kernel(
    const float* __restrict__ Wq, const float* __restrict__ Wk,
    const float* __restrict__ Wv, unsigned short* __restrict__ Wf) {
  const int idx  = blockIdx.x * 256 + threadIdx.x;  // 0 .. 196607
  const int j    = idx & 15;
  const int lane = (idx >> 4) & 31;
  const int nt   = (idx >> 9) & 15;
  const int c    = (idx >> 13) & 7;
  const int w    = idx >> 16;  // 0..2
  const float* W = (w == 0) ? Wq : (w == 1) ? Wk : Wv;
  const int kk = (j < 8) ? (8 * (lane >> 4) + j) : (16 + 8 * (lane >> 4) + (j - 8));
  Wf[idx] = f2bf(W[(c * 32 + kk) * 256 + nt * 16 + (lane & 15)]);
}

// =====================================================================
// Kernel 1: Q/K/V projection.  joint[16384,256] f32 x Wf(bf16 fragments)
//   -> Q  bf16 [B*L,256] row-major
//   -> Kt bf16 [B,256,1024]  (transposed for contiguous QK^T B-frags)
//   -> V  bf16 [B*L,256] row-major
// =====================================================================
__global__ __launch_bounds__(WPB * 32) void qkv_proj_kernel(
    const float* __restrict__ joint, const unsigned short* __restrict__ Wf,
    unsigned short* __restrict__ Qb, unsigned short* __restrict__ Ktb,
    unsigned short* __restrict__ Vb) {
  const int lane  = threadIdx.x & 31;
  const int wave  = threadIdx.x >> 5;
  const int tile  = blockIdx.x * WPB + wave;   // 0..1023 row tiles
  const int rit   = lane & 15;                 // A row within tile
  const int hi    = lane >> 4;
  const int kbase = hi * 8;
  const size_t g0 = (size_t)tile * 16;         // first flattened row

  // A fragments: 16 rows x 256 of joint, f32 -> bf16, 8 chunks of K=32
  BFrag a[8];
#pragma unroll
  for (int c = 0; c < 8; ++c) {
    const float* rp = joint + (g0 + rit) * 256 + c * 32;
#pragma unroll
    for (int j = 0; j < 8; ++j) {
      const int k = a_pair_k(j, kbase);
      a[c].u[j] = f2bf_pk(rp[k], rp[k + 1]);
    }
  }

#pragma unroll
  for (int w = 0; w < 3; ++w) {
    for (int nt = 0; nt < 16; ++nt) {
      v8f acc = {};
#pragma unroll
      for (int c = 0; c < 8; ++c) {
        BFrag bf;  // pre-swizzled: 32 contiguous bytes per lane
        const unsigned short* wp =
            Wf + (size_t)((((w * 8 + c) * 16 + nt) * 32) + lane) * 16;
        bf.q[0] = *(const v4u*)(wp);
        bf.q[1] = *(const v4u*)(wp + 8);
        acc = __builtin_amdgcn_wmma_f32_16x16x32_bf16(
            false, a[c].v, false, bf.v, (short)0, acc, false, false);
      }
      // C/D layout: lane = col (within n-tile), VGPR r = row r + 8*hi
#pragma unroll
      for (int r = 0; r < 8; ++r) {
        const size_t grow = g0 + r + 8 * hi;
        const int    col  = nt * 16 + rit;
        const unsigned short bv = f2bf(acc[r]);
        if (w == 0) {
          Qb[grow * 256 + col] = bv;
        } else if (w == 2) {
          Vb[grow * 256 + col] = bv;
        } else {  // K stored transposed per batch: Kt[b][o][l]
          const size_t bb = grow >> 10, lr = grow & 1023;
          Ktb[(bb * 256 + col) * 1024 + lr] = bv;
        }
      }
    }
  }
}

// =====================================================================
// Kernel 2: fused flash attention with delta-bias and post-softmax mask.
// One wave per (batch, 16-row tile); full O=256 accumulated in registers.
// =====================================================================
__global__ __launch_bounds__(WPB * 32) void attn_kernel(
    const unsigned short* __restrict__ Qb, const unsigned short* __restrict__ Ktb,
    const unsigned short* __restrict__ Vb, const float* __restrict__ delta,
    const int* __restrict__ traj_len, float* __restrict__ out) {
  __shared__ __align__(16) unsigned short plds[WPB][16 * 32];

  const int lane  = threadIdx.x & 31;
  const int wave  = threadIdx.x >> 5;
  const int tile  = blockIdx.x * WPB + wave;   // 0..1023
  const int b     = tile >> 6;
  const int l0    = (tile & 63) * 16;
  const int rit   = lane & 15;
  const int hi    = lane >> 4;
  const int kbase = hi * 8;
  const int tl    = traj_len[b];

  const unsigned short* Q  = Qb  + ((size_t)b * 1024 + l0) * 256;
  const unsigned short* Kt = Ktb + (size_t)b * 256 * 1024;
  const unsigned short* V  = Vb  + (size_t)b * 1024 * 256;
  const float*          Db = delta + (size_t)b * 1024 * 1024 * 2;

  // Resident Q A-fragments (8 x v16bf = 64 VGPRs)
  BFrag qf[8];
#pragma unroll
  for (int c = 0; c < 8; ++c) {
    const unsigned short* rp = Q + rit * 256 + c * 32;
#pragma unroll
    for (int j = 0; j < 8; ++j)
      qf[c].u[j] = *(const unsigned int*)(rp + a_pair_k(j, kbase));
  }

  v8f acc[16];
#pragma unroll
  for (int n = 0; n < 16; ++n) acc[n] = (v8f){};
  float run_max[8], run_sum[8];
#pragma unroll
  for (int r = 0; r < 8; ++r) { run_max[r] = -1e30f; run_sum[r] = 0.0f; }

  unsigned short* lp = plds[wave];  // wave-private 16x32 bf16 P tile

  for (int mc = 0; mc < 32; ++mc) {  // 32 chunks of 32 columns
    const int m0 = mc * 32;
    v8f s[2];
#pragma unroll
    for (int t = 0; t < 2; ++t) {
      const int mt = m0 + t * 16;
      v8f st = {};
#pragma unroll
      for (int c = 0; c < 8; ++c) {
        BFrag kf;  // B frag: lane = m column, 16 E-rows (table-order k) from Kt
#pragma unroll
        for (int j = 0; j < 16; ++j)
          kf.s[j] = Kt[(size_t)(c * 32 + a_pair_k(j >> 1, kbase) + (j & 1)) * 1024 + mt + rit];
        st = __builtin_amdgcn_wmma_f32_16x16x32_bf16(
            false, qf[c].v, false, kf.v, (short)0, st, false, false);
      }
      // add delta bias in-place in C/D layout; stream delta exactly once
#pragma unroll
      for (int r = 0; r < 8; ++r) {
        const size_t row = (size_t)(l0 + r + 8 * hi);
        const float* dp  = Db + (row * 1024 + (mt + rit)) * 2;
        st[r] += dp[0] + dp[1];
        if (t == 0 && mc + 1 < 32) __builtin_prefetch(dp + 64, 0, 0);
      }
      s[t] = st;
    }

    // Online softmax. Row r+8*hi lives in lanes of one 16-lane half;
    // xor masks 1..8 reduce across columns without crossing halves.
    float scale[8];
    bool  need = false;  // any row max increased -> rescale needed
#pragma unroll
    for (int r = 0; r < 8; ++r) {
      float cm = fmaxf(s[0][r], s[1][r]);
#pragma unroll
      for (int m = 1; m <= 8; m <<= 1) cm = fmaxf(cm, __shfl_xor(cm, m, 32));
      need |= (cm > run_max[r]);
      const float nm = fmaxf(run_max[r], cm);
      scale[r]   = __expf(run_max[r] - nm);
      run_max[r] = nm;
      const float p0 = __expf(s[0][r] - nm);
      const float p1 = __expf(s[1][r] - nm);
      float cs = p0 + p1;  // denominator includes ALL m (mask is post-softmax)
#pragma unroll
      for (int m = 1; m <= 8; m <<= 1) cs += __shfl_xor(cs, m, 32);
      run_sum[r] = run_sum[r] * scale[r] + cs;
      // masked P -> LDS (column mask only; numerator drops invalid m)
      const unsigned short pb0 = ((m0 + rit)      < tl) ? f2bf(p0) : (unsigned short)0;
      const unsigned short pb1 = ((m0 + 16 + rit) < tl) ? f2bf(p1) : (unsigned short)0;
      const int prow = r + 8 * hi;
      lp[prow * 32 + rit]      = pb0;
      lp[prow * 32 + 16 + rit] = pb1;
    }

    // Lazy rescale: skip the 128 v_mul block on chunks where no row's max
    // moved (exact: skipped lanes have scale[r] == 1.0). Wave-uniform via
    // ballot so EXEC stays full around the WMMAs either way.
    if (__ballot(need)) {
#pragma unroll
      for (int n = 0; n < 16; ++n)
#pragma unroll
        for (int r = 0; r < 8; ++r) acc[n][r] *= scale[r];
    }

    asm volatile("s_wait_dscnt 0" ::: "memory");  // intra-wave LDS RAW

    // Reread P as A fragment (transpose through LDS)
    BFrag pf;
#pragma unroll
    for (int j = 0; j < 8; ++j)
      pf.u[j] = *(const unsigned int*)(lp + rit * 32 + a_pair_k(j, kbase));

    // PV: accumulate all 16 O-tiles
#pragma unroll
    for (int n = 0; n < 16; ++n) {
      BFrag vf;  // lane = o column, 16 m-rows (table-order k) from V
#pragma unroll
      for (int j = 0; j < 16; ++j)
        vf.s[j] = V[(size_t)(m0 + a_pair_k(j >> 1, kbase) + (j & 1)) * 256 + n * 16 + rit];
      acc[n] = __builtin_amdgcn_wmma_f32_16x16x32_bf16(
          false, pf.v, false, vf.v, (short)0, acc[n], false, false);
    }
  }

  // Epilogue: normalize, zero invalid rows, store f32
#pragma unroll
  for (int r = 0; r < 8; ++r) {
    const int   row = l0 + r + 8 * hi;
    const float inv = (row < tl) ? (1.0f / run_sum[r]) : 0.0f;
#pragma unroll
    for (int n = 0; n < 16; ++n)
      out[((size_t)b * 1024 + row) * 256 + n * 16 + rit] = acc[n][r] * inv;
  }
}

// =====================================================================
extern "C" void kernel_launch(void* const* d_in, const int* in_sizes, int n_in,
                              void* d_out, int out_size, void* d_ws, size_t ws_size,
                              hipStream_t stream) {
  const float* joint = (const float*)d_in[0];
  const float* delta = (const float*)d_in[1];
  const int*   traj  = (const int*)d_in[2];
  const float* Wq    = (const float*)d_in[3];
  const float* Wk    = (const float*)d_in[4];
  const float* Wv    = (const float*)d_in[5];
  float*       out   = (float*)d_out;

  // Workspace: Q (8MB) | Kt (8MB) | V (8MB) | Wf (384KB), all bf16
  unsigned short* Qb  = (unsigned short*)d_ws;
  unsigned short* Ktb = Qb + (size_t)16384 * 256;
  unsigned short* Vb  = Ktb + (size_t)16384 * 256;
  unsigned short* Wfb = Vb + (size_t)16384 * 256;

  wswz_kernel<<<768, 256, 0, stream>>>(Wq, Wk, Wv, Wfb);
  qkv_proj_kernel<<<1024 / WPB, WPB * 32, 0, stream>>>(joint, Wfb, Qb, Ktb, Vb);
  attn_kernel<<<1024 / WPB, WPB * 32, 0, stream>>>(Qb, Ktb, Vb, delta, traj, out);
}
